// MetaPathGNN_24610162606552
// MI455X (gfx1250) — matrix operand order repaired and checked by reading, not compiled, not run on hardware
//
#include <hip/hip_runtime.h>

// MetaPath GNN for MI455X (gfx1250), wave32 + v_wmma_f32_16x16x32_bf16.
// Per layer: memset(agg) -> edge scatter (f32 atomics, L2-resident)
//          -> fused GEMM: relu(agg@wl.T + xin@(w0+w1).T + (bl+b0+b1))
// Weights are pre-split once into bf16 hi/lo in WMMA-B operand layout, so the
// GEMM hot loop does raw b128 loads + WMMAs; only A tiles are split on the fly
// (bf16x3: Ah*Bh + Ah*Bl + Al*Bh, f32 accumulate => ~fp32 accuracy).

typedef __bf16 bf16_t;
typedef __attribute__((ext_vector_type(16))) bf16_t v16bf;
typedef __attribute__((ext_vector_type(8)))  float  v8f;

#define HID 64

// -------- scatter: agg[src[e]] += h[dst[e]] for edges of type `rel` ---------
__global__ void scatter_agg(const int* __restrict__ srcIdx,
                            const int* __restrict__ dstIdx,
                            const int* __restrict__ etype, int rel,
                            const float* __restrict__ h,
                            float* __restrict__ agg,
                            int E, int K, int cShift /* log2(K/4) */)
{
  const int vpr = 1 << cShift;                       // float4 vectors per row
  const int e  = blockIdx.x * (blockDim.x >> cShift) + ((int)threadIdx.x >> cShift);
  if (e >= E) return;
  if (etype[e] != rel) return;
  const int c = ((int)threadIdx.x & (vpr - 1)) << 2; // channel offset
  const int d = dstIdx[e];
  const int s = srcIdx[e];
  const float4 v = *(const float4*)(h + (size_t)d * K + c);
  float* ap = agg + (size_t)s * K + c;
  atomicAdd(ap + 0, v.x);
  atomicAdd(ap + 1, v.y);
  atomicAdd(ap + 2, v.z);
  atomicAdd(ap + 3, v.w);
}

// ---- one-shot weight prep: split wl and (w0+w1) into bf16 hi/lo, packed in
// ---- WMMA B-operand order: idx = (k/32)*2048 + t*512 + lane*16 + j,
// ---- where k = (k/32)*32 + (lane/16)*16 + j, o = t*16 + lane%16.
__global__ void prep_weights(const float* __restrict__ Wl,
                             const float* __restrict__ W0,
                             const float* __restrict__ W1,
                             const float* __restrict__ bl,
                             const float* __restrict__ b0,
                             const float* __restrict__ b1,
                             bf16_t* __restrict__ wlhi, bf16_t* __restrict__ wllo,
                             bf16_t* __restrict__ wshi, bf16_t* __restrict__ wslo,
                             float* __restrict__ bsum, int Kin)
{
  const int i = blockIdx.x * blockDim.x + (int)threadIdx.x;
  if (i < HID) bsum[i] = bl[i] + b0[i] + b1[i];
  if (i >= HID * Kin) return;
  const int chunk = i >> 11;        // 2048 packed elements per 32-wide K chunk
  const int rem   = i & 2047;
  const int t     = rem >> 9;       // column tile 0..3
  const int rem2  = rem & 511;
  const int L     = rem2 >> 4;      // lane 0..31
  const int j     = rem2 & 15;      // element within v16bf
  const int k = chunk * 32 + (L >> 4) * 16 + j;
  const int o = t * 16 + (L & 15);
  const size_t wi = (size_t)o * Kin + k;

  const float wl = Wl[wi];
  bf16_t h = (bf16_t)wl;
  wlhi[i] = h;
  wllo[i] = (bf16_t)(wl - (float)h);

  const float wsv = W0[wi] + W1[wi];     // fused: h@w0.T + xin@w1.T, xin==h
  bf16_t hs = (bf16_t)wsv;
  wshi[i] = hs;
  wslo[i] = (bf16_t)(wsv - (float)hs);
}

// ---- split fp32 -> bf16 hi + bf16 residual --------------------------------
__device__ __forceinline__ void cvt_split16(const float* v, v16bf& hi, v16bf& lo)
{
#pragma unroll
  for (int j = 0; j < 16; ++j) {
    bf16_t h = (bf16_t)v[j];
    hi[j] = h;
    lo[j] = (bf16_t)(v[j] - (float)h);
  }
}

// ------ out = relu(A0@Wl.T + A1@(W0+W1).T + biasSum), one wave = 16x64 ------
__global__ void gemm2_relu(const float* __restrict__ A0, const float* __restrict__ A1,
                           const bf16_t* __restrict__ P0hi, const bf16_t* __restrict__ P0lo,
                           const bf16_t* __restrict__ P1hi, const bf16_t* __restrict__ P1lo,
                           const float* __restrict__ bias,
                           float* __restrict__ out, int nrows, int Kin)
{
  const int lane    = (int)threadIdx.x & 31;
  const int wave    = blockIdx.x * ((int)blockDim.x >> 5) + ((int)threadIdx.x >> 5);
  const int rowBase = wave * 16;
  if (rowBase >= nrows) return;        // wave-uniform: EXEC stays all-ones
  const int half = lane >> 4;
  const int l16  = lane & 15;
  int m = rowBase + l16;               // A-matrix row this lane feeds
  if (m >= nrows) m = nrows - 1;       // safe clamp (stores are guarded)

  v8f acc[4] = {};

  const float*  Amats[2] = {A0, A1};
  const bf16_t* Phi[2]   = {P0hi, P1hi};
  const bf16_t* Plo[2]   = {P0lo, P1lo};

  for (int s = 0; s < 2; ++s) {
    const float*  __restrict__ Am = Amats[s];
    const bf16_t* __restrict__ Bh = Phi[s];
    const bf16_t* __restrict__ Bl = Plo[s];
    for (int kb = 0; kb < Kin; kb += 32) {
      // A tile 16x32: lane element j -> K = kb + half*16 + j
      v16bf ahi, alo;
      {
        float av[16];
        const float* arow = Am + (size_t)m * Kin + kb + half * 16;
        *(float4*)(av + 0)  = *(const float4*)(arow + 0);
        *(float4*)(av + 4)  = *(const float4*)(arow + 4);
        *(float4*)(av + 8)  = *(const float4*)(arow + 8);
        *(float4*)(av + 12) = *(const float4*)(arow + 12);
        cvt_split16(av, ahi, alo);
      }
      const int pbase = (kb >> 5) * 2048 + lane * 16;   // packed B tile base
#pragma unroll
      for (int t = 0; t < 4; ++t) {
        const v16bf bhi = *(const v16bf*)(Bh + pbase + t * 512);
        const v16bf blo = *(const v16bf*)(Bl + pbase + t * 512);
        acc[t] = __builtin_amdgcn_wmma_f32_16x16x32_bf16(false, ahi, false, bhi,
                                                         (short)0, acc[t], false, false);
        acc[t] = __builtin_amdgcn_wmma_f32_16x16x32_bf16(false, ahi, false, blo,
                                                         (short)0, acc[t], false, false);
        acc[t] = __builtin_amdgcn_wmma_f32_16x16x32_bf16(false, alo, false, bhi,
                                                         (short)0, acc[t], false, false);
      }
    }
  }

  // Epilogue: C/D layout -> VGPR r holds row (rowBase + half*8 + r), col l16+16t
#pragma unroll
  for (int t = 0; t < 4; ++t) {
    const int col = t * 16 + l16;
    const float b = bias[col];
#pragma unroll
    for (int r = 0; r < 8; ++r) {
      const int row = rowBase + half * 8 + r;
      if (row < nrows) {
        const float v = acc[t][r] + b;
        out[(size_t)row * HID + col] = v > 0.0f ? v : 0.0f;
      }
    }
  }
}

// ---------------------------------------------------------------------------
extern "C" void kernel_launch(void* const* d_in, const int* in_sizes, int n_in,
                              void* d_out, int out_size, void* d_ws, size_t ws_size,
                              hipStream_t stream)
{
  (void)n_in; (void)out_size; (void)ws_size;

  const float* x  = (const float*)d_in[0];
  const int*   ei = (const int*)d_in[1];   // [2, E] flat
  const int*   et = (const int*)d_in[2];   // [E]
  const int E = in_sizes[2];
  const int N = in_sizes[0] / (2 * HID);
  const int* srcIdx = ei;
  const int* dstIdx = ei + E;

  // d_in[3..20]: per layer l: wl{l}, wbl{l}, w0{l}, wb0{l}, w1{l}, wb1{l}
  const float *W[3][3], *B[3][3];
  for (int l = 0; l < 3; ++l) {
    const int base = 3 + l * 6;
    W[l][0] = (const float*)d_in[base + 0]; B[l][0] = (const float*)d_in[base + 1];
    W[l][1] = (const float*)d_in[base + 2]; B[l][1] = (const float*)d_in[base + 3];
    W[l][2] = (const float*)d_in[base + 4]; B[l][2] = (const float*)d_in[base + 5];
  }

  // ---- workspace carve-out (256B aligned): agg | hA | hB | packed weights --
  char* p = (char*)d_ws;
  auto alloc = [&](size_t bytes) -> char* {
    char* r = p;
    p += (bytes + 255) & ~(size_t)255;
    return r;
  };
  float* agg = (float*)alloc((size_t)N * 2 * HID * sizeof(float));
  float* hA  = (float*)alloc((size_t)N * HID * sizeof(float));
  float* hB  = (float*)alloc((size_t)N * HID * sizeof(float));

  bf16_t *Wlhi[3], *Wllo[3], *Wshi[3], *Wslo[3];
  float  *bsum[3];
  int Kdim[3] = {2 * HID, HID, HID};
  for (int l = 0; l < 3; ++l) {
    const size_t nb = (size_t)HID * Kdim[l] * sizeof(bf16_t);
    Wlhi[l] = (bf16_t*)alloc(nb);
    Wllo[l] = (bf16_t*)alloc(nb);
    Wshi[l] = (bf16_t*)alloc(nb);
    Wslo[l] = (bf16_t*)alloc(nb);
    bsum[l] = (float*)alloc(HID * sizeof(float));
  }
  float* outF = (float*)d_out;

  // ---- one-shot weight prep (tiny; L2-resident afterwards) -----------------
  for (int l = 0; l < 3; ++l) {
    const int K = Kdim[l];
    const int pblocks = (HID * K + 255) / 256;
    prep_weights<<<pblocks, 256, 0, stream>>>(W[l][0], W[l][1], W[l][2],
                                              B[l][0], B[l][1], B[l][2],
                                              Wlhi[l], Wllo[l], Wshi[l], Wslo[l],
                                              bsum[l], K);
  }

  const int waves   = (N + 15) / 16;
  const int gblocks = (waves + 7) / 8;         // 8 waves (256 thr) per block

  const float* hin[3]  = {x, hA, hB};
  float*       hout[3] = {hA, hB, outF};
  for (int l = 0; l < 3; ++l) {
    const int K = Kdim[l];
    const int shift = (K == 128) ? 5 : 4;      // log2(K/4)
    hipMemsetAsync(agg, 0, (size_t)N * K * sizeof(float), stream);
    const int sblocks = (E * (K / 4) + 255) / 256;
    scatter_agg<<<sblocks, 256, 0, stream>>>(srcIdx, dstIdx, et, l,
                                             hin[l], agg, E, K, shift);
    gemm2_relu<<<gblocks, 256, 0, stream>>>(agg, hin[l],
                                            Wlhi[l], Wllo[l], Wshi[l], Wslo[l],
                                            bsum[l], hout[l], N, K);
  }
}
